// Model_1_43413529428145
// MI455X (gfx1250) — matrix operand order repaired and checked by reading, not compiled, not run on hardware
//
#include <hip/hip_runtime.h>
#include <hip/hip_bf16.h>
#include <stddef.h>

// ---------------------------------------------------------------------------
// CDNA5 WMMA (wave32): D(16x16 f32) = A(16x32 bf16) x B(32x16 bf16) + C
// ---------------------------------------------------------------------------
typedef __attribute__((ext_vector_type(16))) __bf16 v16bf;
typedef __attribute__((ext_vector_type(8)))  float  v8f;

#define HID 128        // K dim / hidden width (every GEMM here has K = 128)
#define KSTEPS 4       // 128 / 32

// ---------------------------------------------------------------------------
// Weight pre-pack: W (K=128 x ncols, f32, row-major) -> bf16 B-fragments laid
// out contiguously as frag[tile_n][kstep][lane][16].  ISA 7.12.2 B layout:
//   lane L -> col N = L&15 ; elems 0..7 : K = klo+i, elems 8..15 : K = 16+klo+(i-8)
//   (klo = L<16 ? 0 : 8), all relative to kstep*32.
// Cols >= ncols are zero-padded at pack time (handles the ncols==8 head).
// ---------------------------------------------------------------------------
__global__ void pack_w_bf16(const float* __restrict__ W, __bf16* __restrict__ out,
                            int ncols, int total) {
  const int idx = (int)(blockIdx.x * blockDim.x + threadIdx.x);
  if (idx >= total) return;
  const int i     = idx & 15;
  const int lane  = (idx >> 4) & 31;
  const int kstep = (idx >> 9) & 3;
  const int tn    = idx >> 11;
  const int klo   = (lane & 16) ? 8 : 0;
  const int k     = kstep * 32 + klo + ((i < 8) ? i : (i + 8));
  const int n     = tn * 16 + (lane & 15);
  const float v   = (n < ncols) ? W[(size_t)k * ncols + n] : 0.0f;
  out[idx] = (__bf16)v;
}

// A-fragment: 16 scaled f32 -> bf16, loaded as four float4 (16B aligned:
// klo*4 is 0 or 32 bytes off a 128-float row).
__device__ __forceinline__ v16bf cvt_a_frag(const float* __restrict__ p, float s) {
  const float4 p0 = *(const float4*)(p);
  const float4 p1 = *(const float4*)(p + 4);
  const float4 p2 = *(const float4*)(p + 16);
  const float4 p3 = *(const float4*)(p + 20);
  v16bf a;
  a[0]  = (__bf16)(p0.x * s); a[1]  = (__bf16)(p0.y * s);
  a[2]  = (__bf16)(p0.z * s); a[3]  = (__bf16)(p0.w * s);
  a[4]  = (__bf16)(p1.x * s); a[5]  = (__bf16)(p1.y * s);
  a[6]  = (__bf16)(p1.z * s); a[7]  = (__bf16)(p1.w * s);
  a[8]  = (__bf16)(p2.x * s); a[9]  = (__bf16)(p2.y * s);
  a[10] = (__bf16)(p2.z * s); a[11] = (__bf16)(p2.w * s);
  a[12] = (__bf16)(p3.x * s); a[13] = (__bf16)(p3.y * s);
  a[14] = (__bf16)(p3.z * s); a[15] = (__bf16)(p3.w * s);
  return a;
}

// ---------------------------------------------------------------------------
// Fused dual-GEMM: out[M x ncols] = (A .* rowscale) @ W (+ A2 @ W2) + bias.
// One wave computes 16 rows x (CT*16) cols; A fragment reused across CT WMMAs.
// Fully unrolled: CT=4, DUAL -> 32 v_wmma per wave.
// ---------------------------------------------------------------------------
template <int CT, bool DUAL, bool RELU, bool GUARD>
__global__ void gemm_wmma(const float* __restrict__ A,
                          const float* __restrict__ rowscale,    // nullable
                          const __bf16* __restrict__ Wp,
                          const float* __restrict__ A2,
                          const __bf16* __restrict__ Wp2,
                          const float* __restrict__ bias,
                          float* __restrict__ out,
                          int M, int ncols, int colGroups) {
  const int wave = (int)((blockIdx.x * blockDim.x + threadIdx.x) >> 5);
  const int lane = (int)(threadIdx.x & 31);
  if (wave >= (M >> 4) * colGroups) return;   // wave-uniform (EXEC stays all-1s)
  const int tm   = wave / colGroups;
  const int cg   = wave % colGroups;
  const int row0 = tm << 4;
  const int row  = row0 + (lane & 15);
  const int klo  = (lane & 16) ? 8 : 0;
  const float s  = rowscale ? rowscale[row] : 1.0f;

  const float* aRow  = A + (size_t)row * HID;
  const float* aRow2 = DUAL ? (A2 + (size_t)row * HID) : nullptr;
  const v16bf* bf    = (const v16bf*)Wp  + (size_t)cg * CT * KSTEPS * 32;
  const v16bf* bf2   = DUAL ? ((const v16bf*)Wp2 + (size_t)cg * CT * KSTEPS * 32)
                            : nullptr;

  v8f c[CT] = {};

#pragma unroll
  for (int ks = 0; ks < KSTEPS; ++ks) {
    const v16bf a = cvt_a_frag(aRow + ks * 32 + klo, s);
#pragma unroll
    for (int t = 0; t < CT; ++t) {
      const v16bf b = bf[(t * KSTEPS + ks) * 32 + lane];
      c[t] = __builtin_amdgcn_wmma_f32_16x16x32_bf16(false, a, false, b,
                                                     (short)0, c[t], false, false);
    }
    if (DUAL) {
      const v16bf a2 = cvt_a_frag(aRow2 + ks * 32 + klo, 1.0f);
#pragma unroll
      for (int t = 0; t < CT; ++t) {
        const v16bf b2 = bf2[(t * KSTEPS + ks) * 32 + lane];
        c[t] = __builtin_amdgcn_wmma_f32_16x16x32_bf16(false, a2, false, b2,
                                                       (short)0, c[t], false, false);
      }
    }
  }

  // C/D layout: elem r -> row = row0 + r + 8*(lane>>4), col = base + (lane&15)
  const int mOff = row0 + ((lane >> 4) << 3);
#pragma unroll
  for (int t = 0; t < CT; ++t) {
    const int n = (cg * CT + t) * 16 + (lane & 15);
    if (!GUARD || n < ncols) {
      const float bv = bias[n];
#pragma unroll
      for (int r = 0; r < 8; ++r) {
        float v = c[t][r] + bv;
        if (RELU) v = fmaxf(v, 0.0f);
        out[(size_t)(mOff + r) * ncols + n] = v;
      }
    }
  }
}

// ---------------------------------------------------------------------------
// Helper kernels
// ---------------------------------------------------------------------------
__global__ void zero_f32(float* __restrict__ p, size_t n) {
  size_t i = (size_t)blockIdx.x * blockDim.x + threadIdx.x;
  if (i < n) p[i] = 0.0f;
}

__global__ void edge_degree(const int* __restrict__ dst, float* __restrict__ deg,
                            int nedges) {
  int e = (int)(blockIdx.x * blockDim.x + threadIdx.x);
  if (e < nedges) atomicAdd(&deg[dst[e]], 1.0f);
}

__global__ void recip_deg(float* __restrict__ deg, int n) {
  int i = (int)(blockIdx.x * blockDim.x + threadIdx.x);
  if (i < n) deg[i] = 1.0f / fmaxf(deg[i], 1.0f);
}

// 32 lanes per edge, float4 per lane -> full coalesced 128-ch row per edge.
// x/agg (25.6 MB each) stay resident in the 192 MB L2, so this runs at L2
// gather / L2-atomic throughput, not HBM.
__global__ void edge_scatter_add(const float* __restrict__ xin,
                                 const int* __restrict__ src,
                                 const int* __restrict__ dst,
                                 float* __restrict__ agg, int nedges) {
  const long long t = (long long)blockIdx.x * blockDim.x + threadIdx.x;
  const int e = (int)(t >> 5);
  if (e >= nedges) return;
  const int c = (int)((t & 31) << 2);
  const float4 v = *(const float4*)(xin + (size_t)src[e] * HID + c);
  float* o = agg + (size_t)dst[e] * HID + c;
  atomicAdd(o + 0, v.x);
  atomicAdd(o + 1, v.y);
  atomicAdd(o + 2, v.z);
  atomicAdd(o + 3, v.w);
}

__global__ void graph_pool(const float* __restrict__ h, const int* __restrict__ batch,
                           float* __restrict__ pooled, float* __restrict__ cnt,
                           int nnodes) {
  const long long t = (long long)blockIdx.x * blockDim.x + threadIdx.x;
  const int node = (int)(t >> 5);
  if (node >= nnodes) return;
  const int c = (int)((t & 31) << 2);
  const int g = batch[node];
  const float4 v = *(const float4*)(h + (size_t)node * HID + c);
  float* o = pooled + (size_t)g * HID + c;
  atomicAdd(o + 0, v.x);
  atomicAdd(o + 1, v.y);
  atomicAdd(o + 2, v.z);
  atomicAdd(o + 3, v.w);
  if ((t & 31) == 0) atomicAdd(&cnt[g], 1.0f);
}

__global__ void pool_div(float* __restrict__ pooled, const float* __restrict__ cnt,
                         int total) {
  int i = (int)(blockIdx.x * blockDim.x + threadIdx.x);
  if (i < total) pooled[i] = pooled[i] / fmaxf(cnt[i >> 7], 1.0f);
}

// ---------------------------------------------------------------------------
// Launcher
// ---------------------------------------------------------------------------
static inline int cdiv(long long a, long long b) { return (int)((a + b - 1) / b); }

extern "C" void kernel_launch(void* const* d_in, const int* in_sizes, int n_in,
                              void* d_out, int out_size, void* d_ws, size_t ws_size,
                              hipStream_t stream) {
  (void)n_in; (void)ws_size;
  const float* x      = (const float*)d_in[0];
  const int*   eidx   = (const int*)d_in[1];
  const int*   batch  = (const int*)d_in[2];
  const float* Wf[9]  = {(const float*)d_in[3],  (const float*)d_in[5],
                         (const float*)d_in[6],  (const float*)d_in[8],
                         (const float*)d_in[9],  (const float*)d_in[11],
                         (const float*)d_in[12], (const float*)d_in[14],
                         (const float*)d_in[16]};
  // order: Wrel1, Wroot1, Wrel2, Wroot2, Wrel3, Wroot3, W1, W2, Wo
  const float* brel1 = (const float*)d_in[4];
  const float* brel2 = (const float*)d_in[7];
  const float* brel3 = (const float*)d_in[10];
  const float* b1    = (const float*)d_in[13];
  const float* b2    = (const float*)d_in[15];
  const float* bo    = (const float*)d_in[17];

  const int N = in_sizes[0] / HID;   // 50000 nodes
  const int E = in_sizes[1] / 2;     // 800000 edges
  const int G = out_size / 8;        // 512 graphs
  const int OUTC = 8;
  const int* src = eidx;
  const int* dst = eidx + E;

  // ---- workspace carve-up (f32 region, all 32B-aligned offsets) ----
  float* ws     = (float*)d_ws;
  float* agg    = ws;                           // N*128
  float* deg    = agg + (size_t)N * HID;        // N  (-> 1/max(deg,1))
  float* h1     = deg + N;                      // N*128
  float* h2     = h1 + (size_t)N * HID;         // N*128
  float* pooled = h2 + (size_t)N * HID;         // G*128
  float* cnt    = pooled + (size_t)G * HID;     // G
  float* g1     = cnt + G;                      // G*128
  float* g2     = g1 + (size_t)G * HID;         // G*128
  // bf16 packed-weight region (32B aligned)
  __bf16* wpack = (__bf16*)(g2 + (size_t)G * HID);
  const int FULL = (HID / 16) * KSTEPS * 32 * 16;   // 16384 frags-elems / matrix
  const int OUTP = 1 * KSTEPS * 32 * 16;            // 2048 for Wo (1 col tile)
  __bf16* wp[9];
  for (int i = 0; i < 8; ++i) wp[i] = wpack + (size_t)i * FULL;
  wp[8] = wpack + (size_t)8 * FULL;

  const int TPB = 256;
  const size_t nodeFeats = (size_t)N * HID;

  // ---- pre-pack all weights to bf16 B-fragment layout ----
  for (int i = 0; i < 8; ++i)
    pack_w_bf16<<<cdiv(FULL, TPB), TPB, 0, stream>>>(Wf[i], wp[i], HID, FULL);
  pack_w_bf16<<<cdiv(OUTP, TPB), TPB, 0, stream>>>(Wf[8], wp[8], OUTC, OUTP);

  // ---- degree -> reciprocal (once, reused by all 3 layers) ----
  zero_f32<<<cdiv(N, TPB), TPB, 0, stream>>>(deg, (size_t)N);
  edge_degree<<<cdiv(E, TPB), TPB, 0, stream>>>(dst, deg, E);
  recip_deg<<<cdiv(N, TPB), TPB, 0, stream>>>(deg, N);

  const int scatterBlocks = cdiv((long long)E * 32, TPB);
  const int CG = 2;                                    // 8 col tiles / CT(=4)
  const int layerWaves  = (N >> 4) * CG;
  const int layerBlocks = cdiv((long long)layerWaves * 32, TPB);

  // ---- layer 1: h1 = relu(mean(x) @ Wrel1 + brel1 + x @ Wroot1) ----
  zero_f32<<<cdiv((long long)nodeFeats, TPB), TPB, 0, stream>>>(agg, nodeFeats);
  edge_scatter_add<<<scatterBlocks, TPB, 0, stream>>>(x, src, dst, agg, E);
  gemm_wmma<4, true, true, false><<<layerBlocks, TPB, 0, stream>>>(
      agg, deg, wp[0], x, wp[1], brel1, h1, N, HID, CG);
  // ---- layer 2 ----
  zero_f32<<<cdiv((long long)nodeFeats, TPB), TPB, 0, stream>>>(agg, nodeFeats);
  edge_scatter_add<<<scatterBlocks, TPB, 0, stream>>>(h1, src, dst, agg, E);
  gemm_wmma<4, true, true, false><<<layerBlocks, TPB, 0, stream>>>(
      agg, deg, wp[2], h1, wp[3], brel2, h2, N, HID, CG);
  // ---- layer 3 (no relu) ----
  zero_f32<<<cdiv((long long)nodeFeats, TPB), TPB, 0, stream>>>(agg, nodeFeats);
  edge_scatter_add<<<scatterBlocks, TPB, 0, stream>>>(h2, src, dst, agg, E);
  gemm_wmma<4, true, false, false><<<layerBlocks, TPB, 0, stream>>>(
      agg, deg, wp[4], h2, wp[5], brel3, h1, N, HID, CG);

  // ---- global mean pool over graphs ----
  const size_t poolFeats = (size_t)G * HID;
  zero_f32<<<cdiv((long long)(poolFeats + G), TPB), TPB, 0, stream>>>(
      pooled, poolFeats + G);
  graph_pool<<<cdiv((long long)N * 32, TPB), TPB, 0, stream>>>(h1, batch, pooled,
                                                               cnt, N);
  pool_div<<<cdiv((long long)poolFeats, TPB), TPB, 0, stream>>>(pooled, cnt,
                                                                (int)poolFeats);

  // ---- MLP head ----
  const int mlpWaves  = (G >> 4) * CG;
  const int mlpBlocks = cdiv((long long)mlpWaves * 32, TPB);
  gemm_wmma<4, false, true, false><<<mlpBlocks, TPB, 0, stream>>>(
      pooled, nullptr, wp[6], nullptr, nullptr, b1, g1, G, HID, CG);
  gemm_wmma<4, false, true, false><<<mlpBlocks, TPB, 0, stream>>>(
      g1, nullptr, wp[7], nullptr, nullptr, b2, g2, G, HID, CG);
  const int outWaves  = (G >> 4);
  const int outBlocks = cdiv((long long)outWaves * 32, TPB);
  gemm_wmma<1, false, false, true><<<outBlocks, TPB, 0, stream>>>(
      g2, nullptr, wp[8], nullptr, nullptr, bo, (float*)d_out, G, OUTC, 1);
}
